// MHAttention_72988674228287
// MI455X (gfx1250) — compile-verified, matrix operand order checked by reading, not run on hardware
//
#include <hip/hip_runtime.h>

typedef __bf16 bf16;
typedef __attribute__((ext_vector_type(16))) __bf16 v16bf;
typedef __attribute__((ext_vector_type(8)))  __bf16 v8bf;
typedef __attribute__((ext_vector_type(4)))  __bf16 v4bf;
typedef __attribute__((ext_vector_type(4)))  float  v4f;
typedef __attribute__((ext_vector_type(8)))  float  v8f;

#define WMMA_BF16(a, b, c) \
  __builtin_amdgcn_wmma_f32_16x16x32_bf16(false, (a), false, (b), (short)0, (c), false, false)

// ---------------------------------------------------------------------------
// Load a 16x32 bf16 WMMA operand fragment.
// Element (r, c): r = "row role" (M for A-operand, N for B-operand),
// c = contraction index k in [0,32). Address = base[r*rs + c*cs].
// Per ISA 7.12.2 (16-bit A-matrix 16x32): lane L -> r = L%16, half = L/16;
// VGPR v, elem e: k = 16*(v>=4) + 8*half + 2*(v%4) + e.
// With cs==1 each lane's 16 elements form two contiguous 16-byte runs, which
// the compiler merges into clause'd global_load_b128 pairs (verified in asm).
// ---------------------------------------------------------------------------
__device__ __forceinline__ v16bf load_frag(const bf16* __restrict__ base,
                                           int rs, int cs) {
  const int lane = threadIdx.x & 31;
  const int r  = lane & 15;
  const int hi = lane >> 4;
  v16bf f;
#pragma unroll
  for (int j = 0; j < 16; ++j) {
    const int v = j >> 1;
    const int e = j & 1;
    const int k = ((v & 4) << 2) + (hi << 3) + ((v & 3) << 1) + e;
    f[j] = base[r * rs + k * cs];
  }
  return f;
}

// ---------------------------------------------------------------------------
// fp32 -> bf16 elementwise convert, 4 elements/lane (B128 load, B64 store)
// ---------------------------------------------------------------------------
__global__ void cvt_f32_to_bf16_x4(const v4f* __restrict__ in,
                                   v4bf* __restrict__ out, int n4) {
  int i = blockIdx.x * blockDim.x + threadIdx.x;
  const int stride = gridDim.x * blockDim.x;
  for (; i < n4; i += stride) {
    const v4f v = in[i];
    v4bf o;
#pragma unroll
    for (int j = 0; j < 4; ++j) o[j] = (bf16)v[j];
    out[i] = o;
  }
}

// ---------------------------------------------------------------------------
// C[m,n] = sum_k A[m,k] * W[n,k] + bias[n]   (A @ W^T + b, the nn.Linear)
// A: [M,512] bf16 row-major, W: [512,512] bf16 row-major, bias fp32[512].
// One wave computes a 16x64 tile; all 4 B-fragments are fetched before the
// 4 WMMAs so loads can be staged ahead of the matrix pipe.
// Output modes (uniform branch):
//   0: bf16 [M,512] row-major
//   1: fp32 [M,512] row-major (final d_out)
//   2: bf16 batch-transposed vT[(b*512 + n)*4096 + s], m = b*4096+s — per
//      lane the 8 accumulator rows are 8 consecutive s, one 16B store each.
// ---------------------------------------------------------------------------
__global__ __launch_bounds__(128) void gemm_xwt_bf16(
    const bf16* __restrict__ A, const bf16* __restrict__ W,
    const float* __restrict__ bias, void* __restrict__ Out, int M, int mode) {
  const int K = 512, N = 512, S = 4096;
  const int gw = blockIdx.x * (blockDim.x >> 5) + (threadIdx.x >> 5);
  const int nTiles = N / 64;                 // 8
  const int m0 = (gw / nTiles) * 16;
  const int n0 = (gw % nTiles) * 64;
  if (m0 >= M) return;

  v8f acc[4] = {};
#pragma unroll 2
  for (int k0 = 0; k0 < K; k0 += 32) {
    const v16bf a = load_frag(A + (size_t)m0 * K + k0, K, 1);
    v16bf b[4];
#pragma unroll
    for (int i = 0; i < 4; ++i)
      b[i] = load_frag(W + (size_t)(n0 + 16 * i) * K + k0, K, 1);
#pragma unroll
    for (int i = 0; i < 4; ++i) acc[i] = WMMA_BF16(a, b[i], acc[i]);
  }

  const int lane = threadIdx.x & 31;
  const int nl = lane & 15, hi = lane >> 4;
  float bv[4];
#pragma unroll
  for (int i = 0; i < 4; ++i) bv[i] = bias[n0 + 16 * i + nl];

  // D layout: lane -> column n0+nl (+16*i), VGPR r -> row m0 + r + 8*hi
  const size_t base = (size_t)(m0 + 8 * hi) * N + (size_t)(n0 + nl);
  if (mode == 1) {
    float* __restrict__ p = (float*)Out + base;
#pragma unroll
    for (int i = 0; i < 4; ++i)
#pragma unroll
      for (int r = 0; r < 8; ++r)
        p[(size_t)r * N + 16 * i] = acc[i][r] + bv[i];
  } else if (mode == 0) {
    bf16* __restrict__ p = (bf16*)Out + base;
#pragma unroll
    for (int i = 0; i < 4; ++i)
#pragma unroll
      for (int r = 0; r < 8; ++r)
        p[(size_t)r * N + 16 * i] = (bf16)(acc[i][r] + bv[i]);
  } else {
    // batch-transposed: row = channel n, col = sequence position s
    const int bb = m0 >> 12;                 // m0 / 4096 (tiles never straddle)
    const int s0 = (m0 & (S - 1)) + 8 * hi;  // 8-element aligned
    bf16* __restrict__ p = (bf16*)Out;
#pragma unroll
    for (int i = 0; i < 4; ++i) {
      v8bf pk;
#pragma unroll
      for (int r = 0; r < 8; ++r) pk[r] = (bf16)(acc[i][r] + bv[i]);
      const int n = n0 + 16 * i + nl;
      *(v8bf*)(p + (size_t)(bb * N + n) * S + s0) = pk;   // 16B store
    }
  }
}

// ---------------------------------------------------------------------------
// Attention core (no softmax — faithful to the reference bug):
//   out[q,:] += score(q,k) * v[k,:],  score = mask ? (q.k)/8 : -10000
// Q/K: [B*S, 512] bf16 row-major (head h in cols h*64..h*64+63).
// Vt:  [B*512, 4096] bf16 batch-transposed, so V B-fragments gather with
//      cs==1 and merge into b128 loads like every other operand.
// One block = one (b, 16-row q-tile); wave w handles head w, so the 8 waves
// share the mask tile through L0/L2. Keys streamed in 32-wide tiles:
//   4 WMMAs for scores (K=64), score D-tile -> LDS -> A-fragment re-layout,
//   4 WMMAs for the 16x64 output accumulation. 128 iterations over S=4096.
// ---------------------------------------------------------------------------
__global__ __launch_bounds__(256) void attn_scores_v(
    const bf16* __restrict__ Q, const bf16* __restrict__ Kt,
    const bf16* __restrict__ Vt, const int* __restrict__ mask,
    bf16* __restrict__ O) {
  const int S = 4096, E = 512, HD = 64;
  __shared__ bf16 sS[8][16 * 32];                 // per-wave 16x32 score tile

  const int w    = threadIdx.x >> 5;              // head index
  const int lane = threadIdx.x & 31;
  const int nl = lane & 15, hi = lane >> 4;
  const int qtiles = S / 16;
  const int b  = blockIdx.x / qtiles;
  const int q0 = (blockIdx.x % qtiles) * 16;

  const size_t rowBase = (size_t)b * S * E;
  const bf16* qbase = Q + rowBase + (size_t)q0 * E + w * HD;
  const v16bf qa0 = load_frag(qbase +  0, E, 1);  // q tile 16x64 = 2 A-frags
  const v16bf qa1 = load_frag(qbase + 32, E, 1);

  const int* mrow = mask + (size_t)b * S * S + (size_t)q0 * S;  // [16, S]
  const bf16* vhead = Vt + (size_t)(b * E + w * HD) * S;        // [64, S]
  v8f o[4] = {};

  for (int k0 = 0; k0 < S; k0 += 32) {
#pragma unroll
    for (int t = 0; t < 2; ++t) {
      const bf16* kbase = Kt + rowBase + (size_t)(k0 + 16 * t) * E + w * HD;
      const v16bf kb0 = load_frag(kbase +  0, E, 1);   // B = k^T (hd x 16)
      const v16bf kb1 = load_frag(kbase + 32, E, 1);
      v8f s = {};
      s = WMMA_BF16(qa0, kb0, s);
      s = WMMA_BF16(qa1, kb1, s);
      // scale by 1/sqrt(64), apply mask sentinel, park in LDS as bf16
      const int ncol = 16 * t + nl;
#pragma unroll
      for (int r = 0; r < 8; ++r) {
        const int m = r + 8 * hi;
        const int mv = mrow[(size_t)m * S + (k0 + ncol)];
        const float sv = (mv == 0) ? -10000.0f : s[r] * 0.125f;
        sS[w][m * 32 + ncol] = (bf16)sv;
      }
    }
    // reload the score tile as an A-fragment (in-order DS within this wave)
    const v16bf sf = load_frag(&sS[w][0], 32, 1);
#pragma unroll
    for (int i = 0; i < 4; ++i) {
      // B = vT[d0:d0+16, k0:k0+32]: element (n=d, c=k) at vhead[(16i+n)*S + k]
      const v16bf vbf = load_frag(vhead + (size_t)(16 * i) * S + k0, S, 1);
      o[i] = WMMA_BF16(sf, vbf, o[i]);
    }
  }

  bf16* __restrict__ obase =
      O + rowBase + (size_t)(q0 + 8 * hi) * E + w * HD + nl;
#pragma unroll
  for (int i = 0; i < 4; ++i)
#pragma unroll
    for (int r = 0; r < 8; ++r)
      obase[(size_t)r * E + 16 * i] = (bf16)o[i][r];
}

// ---------------------------------------------------------------------------
extern "C" void kernel_launch(void* const* d_in, const int* in_sizes, int n_in,
                              void* d_out, int out_size, void* d_ws,
                              size_t ws_size, hipStream_t stream) {
  const float* x    = (const float*)d_in[0];
  const int*   mask = (const int*)  d_in[1];
  const float* wq   = (const float*)d_in[2];
  const float* bq   = (const float*)d_in[3];
  const float* wk   = (const float*)d_in[4];
  const float* bk   = (const float*)d_in[5];
  const float* wv   = (const float*)d_in[6];
  const float* bv   = (const float*)d_in[7];
  const float* wo   = (const float*)d_in[8];
  const float* bo   = (const float*)d_in[9];
  float* out = (float*)d_out;

  const int B = 2, S = 4096, E = 512;
  const int N = B * S;                       // 8192 rows

  // workspace carve-up (bf16 elements), total ~42 MiB
  bf16* ws  = (bf16*)d_ws;
  bf16* xbf = ws;                            // N*E
  bf16* wqb = xbf + (size_t)N * E;           // E*E each
  bf16* wkb = wqb + (size_t)E * E;
  bf16* wvb = wkb + (size_t)E * E;
  bf16* wob = wvb + (size_t)E * E;
  bf16* qb  = wob + (size_t)E * E;           // N*E each
  bf16* kb  = qb  + (size_t)N * E;
  bf16* vtb = kb  + (size_t)N * E;           // V, batch-transposed [B*E, S]
  bf16* ab  = vtb + (size_t)N * E;           // attention output, pre-projection

  // 1) fp32 -> bf16 operands (vectorized x4)
  cvt_f32_to_bf16_x4<<<1024, 256, 0, stream>>>((const v4f*)x,  (v4bf*)xbf,
                                               N * E / 4);
  cvt_f32_to_bf16_x4<<<64,   256, 0, stream>>>((const v4f*)wq, (v4bf*)wqb,
                                               E * E / 4);
  cvt_f32_to_bf16_x4<<<64,   256, 0, stream>>>((const v4f*)wk, (v4bf*)wkb,
                                               E * E / 4);
  cvt_f32_to_bf16_x4<<<64,   256, 0, stream>>>((const v4f*)wv, (v4bf*)wvb,
                                               E * E / 4);
  cvt_f32_to_bf16_x4<<<64,   256, 0, stream>>>((const v4f*)wo, (v4bf*)wob,
                                               E * E / 4);

  // 2) q,k,v projections: (N/16)*(512/64)=4096 waves, 4 waves/block
  const int gemmBlocks = (N / 16) * (E / 64) / 4;   // 1024
  gemm_xwt_bf16<<<gemmBlocks, 128, 0, stream>>>(xbf, wqb, bq, qb,  N, 0);
  gemm_xwt_bf16<<<gemmBlocks, 128, 0, stream>>>(xbf, wkb, bk, kb,  N, 0);
  gemm_xwt_bf16<<<gemmBlocks, 128, 0, stream>>>(xbf, wvb, bv, vtb, N, 2);

  // 3) masked-score @ v, streamed (no [B,H,S,S] materialization)
  attn_scores_v<<<B * (S / 16), 256, 0, stream>>>(qb, kb, vtb, mask, ab);

  // 4) output projection, fp32 store to d_out
  gemm_xwt_bf16<<<gemmBlocks, 128, 0, stream>>>(ab, wob, bo, out, N, 1);
}
